// SAModule_4801773437698
// MI455X (gfx1250) — matrix operand order repaired and checked by reading, not compiled, not run on hardware
//
#include <hip/hip_runtime.h>
#include <hip/hip_bf16.h>

#define NB      4
#define NPTS    16384
#define CFEAT   32
#define M_PER   4096
#define M_TOT   (NB*M_PER)
#define KNB     33          // K+1 neighbors
#define R2      0.04f       // r^2 = 0.2^2
#define DIN     35          // C + 3
#define DINPAD  64
#define H1DIM   128
#define H2DIM   256
#define H3DIM   256

typedef __attribute__((ext_vector_type(16))) _Float16 v16h;
typedef __attribute__((ext_vector_type(8)))  _Float16 v8h;
typedef __attribute__((ext_vector_type(8)))  float    v8f;

__device__ __forceinline__ v8f wmma16(v16h a, v16h b, v8f c) {
  // D = A(16x32 f16) * B(32x16 f16) + C(16x16 f32)
  return __builtin_amdgcn_wmma_f32_16x16x32_f16(
      /*neg_a=*/false, a, /*neg_b=*/false, b,
      /*c_mod=*/(short)0, c, /*reuse_a=*/false, /*reuse_b=*/false);
}

__device__ __forceinline__ v16h frag_join(v8h lo, v8h hi) {
  union { v16h v; v8h h[2]; } u; u.h[0] = lo; u.h[1] = hi; return u.v;
}

// A fragment (16x32 f16) from row-major LDS base[rows][ldk] (ISA 16-bit A layout).
__device__ __forceinline__ v16h load_a_frag(const _Float16* base, int ldk,
                                            int kbase, int lane) {
  int m = lane & 15, hf = lane >> 4;
  const _Float16* p0 = base + m * ldk + kbase + hf * 8;
  v8h lo = *(const v8h*)p0;
  v8h hi = *(const v8h*)(p0 + 16);
  return frag_join(lo, hi);
}

// B fragment (32x16 f16) from N-major LDS Wt[N][ldk]: lane reads k 8hf..8hf+7 and 16+8hf..16+8hf+7.
__device__ __forceinline__ v16h load_b_frag(const _Float16* Wt, int ldk,
                                            int nbase, int kbase, int lane) {
  int n = lane & 15, hf = lane >> 4;
  const _Float16* p = Wt + (size_t)(nbase + n) * ldk + kbase + hf * 8;
  return frag_join(*(const v8h*)p, *(const v8h*)(p + 16));
}

// ---------------------------------------------------------------------------
// Tensor Data Mover: 2-D f16 tile, global -> LDS. D# per ISA cdna5 ch.8.
// Guarded: falls back to manual LDS fills if the builtin is absent.
// ---------------------------------------------------------------------------
#if defined(__HIP_DEVICE_COMPILE__) && defined(__has_builtin)
# if __has_builtin(__builtin_amdgcn_tensor_load_to_lds) && \
     __has_builtin(__builtin_amdgcn_s_wait_tensorcnt)
#  define HAVE_TDM 1
# endif
#endif

#ifdef HAVE_TDM
typedef unsigned int u32x4 __attribute__((ext_vector_type(4)));
typedef int          i32x4 __attribute__((ext_vector_type(4)));
typedef int          i32x8 __attribute__((ext_vector_type(8)));

// tile_w elements per row (2B each), tile_h rows, row_stride in elements.
__device__ __forceinline__ void tdm_load_2d_f16(unsigned lds_off, const void* gptr,
                                                unsigned tile_w, unsigned tile_h,
                                                unsigned row_stride) {
  unsigned long long ga = (unsigned long long)gptr;
  u32x4 g0;
  g0[0] = 1u;                                         // count=1, user descriptor
  g0[1] = lds_off;                                    // lds_addr (bytes)
  g0[2] = (unsigned)(ga & 0xffffffffu);               // global_addr[31:0]
  g0[3] = (unsigned)((ga >> 32) & 0x01ffffffu) | (2u << 30); // addr[56:32], type=2
  i32x8 g1;
  g1[0] = (int)(1u << 16);                            // data_size=1 (2 bytes)
  g1[1] = (int)((tile_w & 0xffffu) << 16);            // tensor_dim0[15:0]
  g1[2] = (int)(((tile_w >> 16) & 0xffffu) | ((tile_h & 0xffffu) << 16)); // dim0 hi | dim1 lo
  g1[3] = (int)(((tile_h >> 16) & 0xffffu) | ((tile_w & 0xffffu) << 16)); // dim1 hi | tile_dim0
  g1[4] = (int)(tile_h & 0xffffu);                    // tile_dim1 (tile_dim2=0)
  g1[5] = (int)row_stride;                            // tensor_dim0_stride[31:0]
  g1[6] = 0;
  g1[7] = 0;
  i32x4 z4 = {0, 0, 0, 0};
#if __clang_major__ >= 23
  i32x8 z8 = {0, 0, 0, 0, 0, 0, 0, 0};
  __builtin_amdgcn_tensor_load_to_lds(g0, g1, z4, z4, z8, 0);
#else
  __builtin_amdgcn_tensor_load_to_lds(g0, g1, z4, z4, 0);
#endif
}
#endif  // HAVE_TDM

// ---------------------------------------------------------------------------
// 0) One-shot weight prep: f32 [K][N] -> f16 N-major [N][Kpad] panels in ws,
//    so the conv kernel can pull them with a single TDM DMA per panel.
// ---------------------------------------------------------------------------
__global__ __launch_bounds__(256) void wprep_kernel(
    const float* __restrict__ lw1, const float* __restrict__ lw2,
    _Float16* __restrict__ wt1, _Float16* __restrict__ wt2) {
  int i = blockIdx.x * 256 + threadIdx.x;
  if (i < H1DIM * DINPAD) {            // wt1[n][k], k padded 35->64
    int n = i >> 6, k = i & 63;
    wt1[i] = (k < DIN) ? (_Float16)lw1[k * H1DIM + n] : (_Float16)0.f;
  }
  int j = i - H1DIM * DINPAD;
  if (j >= 0 && j < H2DIM * H1DIM) {   // wt2[n][k]
    int n = j >> 7, k = j & 127;
    wt2[j] = (_Float16)lw2[k * H2DIM + n];
  }
}

// ---------------------------------------------------------------------------
// 1) Farthest point sampling: one block per cloud, points in registers.
// ---------------------------------------------------------------------------
__global__ __launch_bounds__(1024) void fps_kernel(const float* __restrict__ pos,
                                                   int* __restrict__ idx_out) {
  const int b = blockIdx.x;
  const float* p = pos + (size_t)b * NPTS * 3;
  const int t = threadIdx.x;
  __shared__ float s_d[32];
  __shared__ int   s_i[32];
  __shared__ float s_cur[3];

  float px[16], py[16], pz[16], mind[16];
  float cx = p[0], cy = p[1], cz = p[2];
#pragma unroll
  for (int j = 0; j < 16; ++j) {
    int i = t + j * 1024;
    px[j] = p[i * 3 + 0]; py[j] = p[i * 3 + 1]; pz[j] = p[i * 3 + 2];
    float dx = px[j] - cx, dy = py[j] - cy, dz = pz[j] - cz;
    mind[j] = dx * dx + dy * dy + dz * dz;
  }
  if (t == 0) idx_out[b * M_PER] = b * NPTS;

  const int lane = t & 31, wid = t >> 5;
  for (int m = 1; m < M_PER; ++m) {
    float bd = -1.f; int bi = 0x7fffffff;
#pragma unroll
    for (int j = 0; j < 16; ++j) {
      int i = t + j * 1024;
      if (mind[j] > bd || (mind[j] == bd && i < bi)) { bd = mind[j]; bi = i; }
    }
#pragma unroll
    for (int off = 16; off > 0; off >>= 1) {
      float od = __shfl_down(bd, off, 32);
      int   oi = __shfl_down(bi, off, 32);
      if (od > bd || (od == bd && oi < bi)) { bd = od; bi = oi; }
    }
    if (lane == 0) { s_d[wid] = bd; s_i[wid] = bi; }
    __syncthreads();
    if (t == 0) {
      float wd = s_d[0]; int wi = s_i[0];
      for (int k = 1; k < 32; ++k)
        if (s_d[k] > wd || (s_d[k] == wd && s_i[k] < wi)) { wd = s_d[k]; wi = s_i[k]; }
      idx_out[b * M_PER + m] = b * NPTS + wi;
      s_cur[0] = p[wi * 3 + 0]; s_cur[1] = p[wi * 3 + 1]; s_cur[2] = p[wi * 3 + 2];
    }
    __syncthreads();
    cx = s_cur[0]; cy = s_cur[1]; cz = s_cur[2];
#pragma unroll
    for (int j = 0; j < 16; ++j) {
      float dx = px[j] - cx, dy = py[j] - cy, dz = pz[j] - cz;
      mind[j] = fminf(mind[j], dx * dx + dy * dy + dz * dz);
    }
  }
}

// ---------------------------------------------------------------------------
// 2) 33-NN per centroid: d^2 in registers, 33 tie-broken argmin rounds
//    (matches lax.top_k ordering), radius-validity bitmask.
// ---------------------------------------------------------------------------
__global__ __launch_bounds__(256) void knn_kernel(
    const float* __restrict__ pos, const int* __restrict__ idx,
    const int* __restrict__ batch,
    int* __restrict__ nbr, unsigned long long* __restrict__ vmask,
    float* __restrict__ pos_dst, int* __restrict__ batch_out) {
  const int c = blockIdx.x;
  const int t = threadIdx.x;
  const int b = c / M_PER;
  const int gc = idx[c];
  const float cx = pos[gc * 3 + 0], cy = pos[gc * 3 + 1], cz = pos[gc * 3 + 2];
  const float* p = pos + (size_t)b * NPTS * 3;

  float d2loc[64];
#pragma unroll
  for (int j = 0; j < 64; ++j) {
    int i = t + j * 256;
    float dx = p[i * 3 + 0] - cx, dy = p[i * 3 + 1] - cy, dz = p[i * 3 + 2] - cz;
    d2loc[j] = dx * dx + dy * dy + dz * dz;
  }

  __shared__ float s_d[8];
  __shared__ int   s_i[8];
  __shared__ float s_seld;
  __shared__ int   s_seli;
  float last_d = -1.f; int last_i = -1;
  unsigned long long mask = 0ull;
  const int lane = t & 31, wid = t >> 5;

  for (int k = 0; k < KNB; ++k) {
    float bd = 3.4e38f; int bi = 0x7fffffff;
#pragma unroll
    for (int j = 0; j < 64; ++j) {
      int i = t + j * 256;
      bool after = (d2loc[j] > last_d) || (d2loc[j] == last_d && i > last_i);
      if (after && (d2loc[j] < bd || (d2loc[j] == bd && i < bi))) {
        bd = d2loc[j]; bi = i;
      }
    }
#pragma unroll
    for (int off = 16; off > 0; off >>= 1) {
      float od = __shfl_down(bd, off, 32);
      int   oi = __shfl_down(bi, off, 32);
      if (od < bd || (od == bd && oi < bi)) { bd = od; bi = oi; }
    }
    if (lane == 0) { s_d[wid] = bd; s_i[wid] = bi; }
    __syncthreads();
    if (t == 0) {
      float wd = s_d[0]; int wi = s_i[0];
      for (int w = 1; w < 8; ++w)
        if (s_d[w] < wd || (s_d[w] == wd && s_i[w] < wi)) { wd = s_d[w]; wi = s_i[w]; }
      nbr[c * KNB + k] = b * NPTS + wi;
      if (wd <= R2) mask |= (1ull << k);
      s_seld = wd; s_seli = wi;
    }
    __syncthreads();
    last_d = s_seld; last_i = s_seli;
  }
  if (t == 0) {
    vmask[c] = mask;
    pos_dst[c * 3 + 0] = cx; pos_dst[c * 3 + 1] = cy; pos_dst[c * 3 + 2] = cz;
    batch_out[c] = batch[gc];
  }
}

// ---------------------------------------------------------------------------
// 3) Fused PointNetConv: per centroid A=[48x64] edge features (33 valid),
//    h1=relu(A@W1) 48x128, h2=relu(h1@W2) 48x256, masked row-max -> agg (f16).
//    f16 weight panels DMAed into LDS by the TDM (wave 0), fallback: copy.
// ---------------------------------------------------------------------------
__global__ __launch_bounds__(256) void conv_kernel(
    const float* __restrict__ x, const float* __restrict__ pos,
    const int* __restrict__ idx, const int* __restrict__ nbr,
    const unsigned long long* __restrict__ vmask,
    const _Float16* __restrict__ wt1, const float* __restrict__ lb1,
    const _Float16* __restrict__ wt2, const float* __restrict__ lb2,
    _Float16* __restrict__ agg) {
  __shared__ _Float16 Wt1[H1DIM][DINPAD];   // 16 KB (N-major)
  __shared__ _Float16 Wt2[H2DIM][H1DIM];    // 64 KB (N-major)
  __shared__ float    sb1[H1DIM];
  __shared__ float    sb2[H2DIM];
  __shared__ _Float16 Atile[48][DINPAD];    // 6 KB
  __shared__ _Float16 H1s[48][H1DIM];       // 12 KB

  const int t = threadIdx.x, lane = t & 31, w = t >> 5;

#ifdef HAVE_TDM
  if (t < 32) {   // wave 0 issues both weight-panel DMAs, then drains TENSORcnt
    tdm_load_2d_f16((unsigned)(unsigned long long)(void*)&Wt1[0][0], wt1,
                    DINPAD, H1DIM, DINPAD);
    tdm_load_2d_f16((unsigned)(unsigned long long)(void*)&Wt2[0][0], wt2,
                    H1DIM, H2DIM, H1DIM);
  }
#else
  for (int i = t; i < H1DIM * DINPAD; i += 256) (&Wt1[0][0])[i] = wt1[i];
  for (int i = t; i < H2DIM * H1DIM; i += 256) (&Wt2[0][0])[i] = wt2[i];
#endif
  for (int i = t; i < H1DIM; i += 256) sb1[i] = lb1[i];
  for (int i = t; i < H2DIM; i += 256) sb2[i] = lb2[i];
#ifdef HAVE_TDM
  if (t < 32) __builtin_amdgcn_s_wait_tensorcnt(0);
#endif
  __syncthreads();

  for (int cl = 0; cl < 32; ++cl) {
    const int c = blockIdx.x * 32 + cl;
    const int gc = idx[c];
    const float pcx = pos[gc * 3 + 0], pcy = pos[gc * 3 + 1], pcz = pos[gc * 3 + 2];

    // Edge-feature tile: rows = neighbors (pad 33->48), cols = [x_j | pos_j-pos_i | 0].
    for (int i = t; i < 48 * DINPAD; i += 256) {
      int row = i >> 6, k = i & 63;
      _Float16 v = (_Float16)0.f;
      if (row < KNB && k < DIN) {
        int col = nbr[c * KNB + row];
        if (k < CFEAT) {
          v = (_Float16)x[(size_t)col * CFEAT + k];
        } else {
          float pc = (k == 32) ? pcx : ((k == 33) ? pcy : pcz);
          v = (_Float16)(pos[col * 3 + (k - 32)] - pc);
        }
      }
      Atile[row][k] = v;
    }
    __syncthreads();

    // Layer 1: 48x64 @ 64x128; wave w owns N-tile w.
    v8f acc1[3] = {};
#pragma unroll
    for (int ks = 0; ks < 2; ++ks) {
      v16h bf = load_b_frag(&Wt1[0][0], DINPAD, w * 16, ks * 32, lane);
#pragma unroll
      for (int mt = 0; mt < 3; ++mt) {
        v16h af = load_a_frag(&Atile[mt * 16][0], DINPAD, ks * 32, lane);
        acc1[mt] = wmma16(af, bf, acc1[mt]);
      }
    }
    {
      int n = w * 16 + (lane & 15);
      float bias = sb1[n];
      int rofs = (lane >> 4) << 3;
#pragma unroll
      for (int mt = 0; mt < 3; ++mt)
#pragma unroll
        for (int r = 0; r < 8; ++r) {
          float v = acc1[mt][r] + bias;
          H1s[mt * 16 + r + rofs][n] = (_Float16)(v > 0.f ? v : 0.f);
        }
    }
    __syncthreads();

    // Layer 2: 48x128 @ 128x256; wave w owns N-tiles 2w, 2w+1. Then masked max.
    v8f acc2[3][2] = {};
#pragma unroll
    for (int ks = 0; ks < 4; ++ks) {
      v16h af[3];
#pragma unroll
      for (int mt = 0; mt < 3; ++mt)
        af[mt] = load_a_frag(&H1s[mt * 16][0], H1DIM, ks * 32, lane);
#pragma unroll
      for (int nt = 0; nt < 2; ++nt) {
        v16h bf = load_b_frag(&Wt2[0][0], H1DIM, (2 * w + nt) * 16, ks * 32, lane);
#pragma unroll
        for (int mt = 0; mt < 3; ++mt)
          acc2[mt][nt] = wmma16(af[mt], bf, acc2[mt][nt]);
      }
    }
    const unsigned long long mk = vmask[c];
    const int rofs = (lane >> 4) << 3;
#pragma unroll
    for (int nt = 0; nt < 2; ++nt) {
      int n = (2 * w + nt) * 16 + (lane & 15);
      float bias = sb2[n];
      float m = -3.4e38f;
#pragma unroll
      for (int mt = 0; mt < 3; ++mt)
#pragma unroll
        for (int r = 0; r < 8; ++r) {
          int row = mt * 16 + r + rofs;
          float v = acc2[mt][nt][r] + bias;
          v = v > 0.f ? v : 0.f;
          bool valid = (row < KNB) && ((mk >> row) & 1ull);
          m = valid ? fmaxf(m, v) : m;
        }
      m = fmaxf(m, __shfl_xor(m, 16, 32));  // combine row halves (M vs M+8)
      if (lane < 16) agg[(size_t)c * H2DIM + n] = (_Float16)m;
    }
    __syncthreads();
  }
}

// ---------------------------------------------------------------------------
// 4) Global layer: out = relu(agg[16384,256] @ gw1[256,256] + gb1).
//    Whole 128-row A panel DMAed into LDS via TDM; Bt panel LDS-resident;
//    barrier-free K loop (8 k-steps x 8 N-tiles per wave).
// ---------------------------------------------------------------------------
__global__ __launch_bounds__(256) void gemm3_kernel(
    const _Float16* __restrict__ agg, const float* __restrict__ gw1,
    const float* __restrict__ gb1, float* __restrict__ out) {
  __shared__ _Float16 As[128][H2DIM];   // 64 KB, row-major
  __shared__ _Float16 Bt[128][H3DIM];   // 64 KB, N-major (n rel. to col0)
  const int t = threadIdx.x, lane = t & 31, w = t >> 5;
  const int row0 = blockIdx.x * 128, col0 = blockIdx.y * 128;

#ifdef HAVE_TDM
  if (t < 32)
    tdm_load_2d_f16((unsigned)(unsigned long long)(void*)&As[0][0],
                    agg + (size_t)row0 * H2DIM, H2DIM, 128, H2DIM);
#else
  for (int i = t; i < 128 * H2DIM; i += 256) {
    int r = i >> 8, kk = i & 255;
    As[r][kk] = agg[(size_t)(row0 + r) * H2DIM + kk];
  }
#endif
  for (int i = t; i < 128 * H3DIM; i += 256) {   // coalesced on n
    int n = i & 127, kk = i >> 7;
    Bt[n][kk] = (_Float16)gw1[kk * H3DIM + col0 + n];
  }
#ifdef HAVE_TDM
  if (t < 32) __builtin_amdgcn_s_wait_tensorcnt(0);
#endif
  __syncthreads();

  v8f acc[8] = {};
#pragma unroll
  for (int ks = 0; ks < 8; ++ks) {
    v16h af = load_a_frag(&As[w * 16][0], H2DIM, ks * 32, lane);
#pragma unroll
    for (int nt = 0; nt < 8; ++nt) {
      v16h bf = load_b_frag(&Bt[0][0], H3DIM, nt * 16, ks * 32, lane);
      acc[nt] = wmma16(af, bf, acc[nt]);
    }
  }

  const int rofs = (lane >> 4) << 3;
#pragma unroll
  for (int nt = 0; nt < 8; ++nt) {
    int col = col0 + nt * 16 + (lane & 15);
    float bias = gb1[col];
#pragma unroll
    for (int r = 0; r < 8; ++r) {
      int row = row0 + w * 16 + r + rofs;
      float v = acc[nt][r] + bias;
      out[(size_t)row * H3DIM + col] = v > 0.f ? v : 0.f;
    }
  }
}

// ---------------------------------------------------------------------------
extern "C" void kernel_launch(void* const* d_in, const int* in_sizes, int n_in,
                              void* d_out, int out_size, void* d_ws, size_t ws_size,
                              hipStream_t stream) {
  const float* x     = (const float*)d_in[0];
  const float* pos   = (const float*)d_in[1];
  const int*   batch = (const int*)d_in[2];
  const float* lw1   = (const float*)d_in[3];
  const float* lb1   = (const float*)d_in[4];
  const float* lw2   = (const float*)d_in[5];
  const float* lb2   = (const float*)d_in[6];
  const float* gw1   = (const float*)d_in[7];
  const float* gb1   = (const float*)d_in[8];

  float* out      = (float*)d_out;                       // [16384,256]
  float* pos_dst  = out + (size_t)M_TOT * H3DIM;         // [16384,3]
  int*   batch_o  = (int*)(pos_dst + (size_t)M_TOT * 3); // [16384]

  char* ws = (char*)d_ws;
  int* idx                  = (int*)(ws + 0);                      //  64 KB
  int* nbr                  = (int*)(ws + 65536);                  //  2.06 MB
  unsigned long long* vmask = (unsigned long long*)(ws + 2228224); // 128 KB
  _Float16* agg             = (_Float16*)(ws + 2359296);           //  8 MB
  _Float16* wt1             = (_Float16*)(ws + 10747904);          //  16 KB f16 W1^T
  _Float16* wt2             = (_Float16*)(ws + 10764288);          //  64 KB f16 W2^T

  wprep_kernel<<<(H1DIM * DINPAD + H2DIM * H1DIM + 255) / 256, 256, 0, stream>>>(
      lw1, lw2, wt1, wt2);
  fps_kernel<<<NB, 1024, 0, stream>>>(pos, idx);
  knn_kernel<<<M_TOT, 256, 0, stream>>>(pos, idx, batch, nbr, vmask, pos_dst, batch_o);
  conv_kernel<<<M_TOT / 32, 256, 0, stream>>>(x, pos, idx, nbr, vmask,
                                              wt1, lb1, wt2, lb2, agg);
  gemm3_kernel<<<dim3(M_TOT / 128, H3DIM / 128), 256, 0, stream>>>(agg, gw1, gb1, out);
}
